// GAT_unit_81913616269562
// MI455X (gfx1250) — compile-verified, matrix operand order checked by reading, not run on hardware
//
#include <hip/hip_runtime.h>
#include <hip/hip_bf16.h>

// ---- problem constants (ST-GCN GAT unit) ----
constexpr int N_ = 16, K_ = 3, C_ = 256, T_ = 128, V_ = 50, H_ = 4;
constexpr int CT = C_ * T_;      // 32768 rows per batch in the aggregation GEMM
constexpr int VP = 64;           // V padded to 64 for WMMA (2 x K=32 steps, 4 x N=16 tiles)
constexpr float NEG_SLOPE = 0.2f;
constexpr float BN_EPS = 1e-5f;

typedef __attribute__((ext_vector_type(16))) __bf16 v16bf;
typedef __attribute__((ext_vector_type(8)))  __bf16 v8bf;
typedef __attribute__((ext_vector_type(8)))  float  v8f;

// ---------------------------------------------------------------------------
// k0: zero BN accumulators (graph-capture safe, re-run every call)
// ---------------------------------------------------------------------------
__global__ void k0_zero(float* __restrict__ bnSum, float* __restrict__ bnSumSq) {
    const int i = threadIdx.x;
    if (i < C_) { bnSum[i] = 0.f; bnSumSq[i] = 0.f; }
}

// ---------------------------------------------------------------------------
// k1: xs[n,c,t,v] = sum_k x[n,k,c,t,v]   (single read of the 315MB tensor)
//     gsum[n,c,v] = sum_t xs[n,c,t,v]    (folded t-reduction for the fp path)
// block = (n*C + c), threads (64 v-lanes, 4 t-rows)
// ---------------------------------------------------------------------------
__global__ void k1_reduce(const float* __restrict__ x,
                          float* __restrict__ xs, float* __restrict__ gsum) {
    const int nc = blockIdx.x;            // n*C + c
    const int n  = nc / C_, c = nc % C_;
    const int vt = threadIdx.x;           // 0..63
    const int tt = threadIdx.y;           // 0..3
    __shared__ float red[4][64];
    float acc = 0.f;
    if (vt < V_) {
        const size_t b0 = (size_t)((n * (K_*C_) + 0*C_ + c) * T_) * V_;
        const size_t b1 = (size_t)((n * (K_*C_) + 1*C_ + c) * T_) * V_;
        const size_t b2 = (size_t)((n * (K_*C_) + 2*C_ + c) * T_) * V_;
        for (int t = tt; t < T_; t += 4) {
            const int o = t * V_ + vt;
            const float s = x[b0 + o] + x[b1 + o] + x[b2 + o];
            xs[(size_t)(nc * T_ + t) * V_ + vt] = s;
            acc += s;
        }
    }
    red[tt][vt] = acc;
    __syncthreads();
    if (tt == 0 && vt < V_)
        gsum[nc * V_ + vt] = red[0][vt] + red[1][vt] + red[2][vt] + red[3][vt];
}

// ---------------------------------------------------------------------------
// k2: fp[n,d,v] = (1/(K*T)) * sum_c W[d,c]*gsum[n,c,v] + b[d]
//     (the 13.4 GFLOP conv collapsed by linearity to a 0.1 GFLOP map)
// ---------------------------------------------------------------------------
__global__ void k2_fp(const float* __restrict__ gsum, const float* __restrict__ Ww,
                      const float* __restrict__ Wb, float* __restrict__ fp) {
    const int nd = blockIdx.x;            // n*C + d
    const int n = nd / C_, d = nd % C_;
    const int v = threadIdx.x;
    if (v >= V_) return;
    float s = 0.f;
    for (int c = 0; c < C_; ++c)
        s += Ww[d * C_ + c] * gsum[(n * C_ + c) * V_ + v];
    fp[nd * V_ + v] = s * (1.0f / (float)(K_ * T_)) + Wb[d];
}

// ---------------------------------------------------------------------------
// k3: pairwise leaky-relu GAT scores + mask + softmax -> TRANSPOSED zero-padded
//     64x64 bf16 attention: attnT[n][w][v] = softmax_w(scores[v,w]).
//     Transposed layout makes the WMMA B fragment two contiguous 16B LDS loads.
// ---------------------------------------------------------------------------
__global__ void k3_attn(const float* __restrict__ fp, const float* __restrict__ att,
                        const int* __restrict__ mask, __bf16* __restrict__ attnT) {
    const int n = blockIdx.x;
    const int tid = threadIdx.x;          // 256 threads
    __shared__ float fps[C_ * V_];        // 51.2 KB
    __shared__ float att_s[C_];
    __shared__ float sc[V_ * 52];
    for (int i = tid; i < C_ * V_; i += 256) fps[i] = fp[n * C_ * V_ + i];
    if (tid < C_) att_s[tid] = att[tid];
    for (int i = tid; i < VP * VP; i += 256) attnT[n * VP * VP + i] = (__bf16)0.0f;
    __syncthreads();
    for (int p = tid; p < V_ * V_; p += 256) {
        const int v = p / V_, w = p % V_;
        float s = 0.f;
        for (int d = 0; d < C_; ++d) {
            float t = fps[d * V_ + v] + fps[d * V_ + w];
            t = (t >= 0.f) ? t : NEG_SLOPE * t;
            s += att_s[d] * t;
        }
        if (mask[v * V_ + w] == 0) s = -1e9f;
        sc[v * 52 + w] = s;
    }
    __syncthreads();
    if (tid < V_) {
        const int v = tid;
        float mx = -3.4e38f;
        for (int w = 0; w < V_; ++w) mx = fmaxf(mx, sc[v * 52 + w]);
        float sum = 0.f;
        for (int w = 0; w < V_; ++w) sum += __expf(sc[v * 52 + w] - mx);
        const float is = 1.0f / sum;
        for (int w = 0; w < V_; ++w)   // transposed store: [w][v]
            attnT[n * VP * VP + w * VP + v] = (__bf16)(__expf(sc[v * 52 + w] - mx) * is);
    }
}

// ---------------------------------------------------------------------------
// k4: out[n,c,t,w] = sum_v xs[n,c,t,v] * attn[n,v,w] via v_wmma_f32_16x16x32_bf16
//     block = (n, c): 128 rows (t) x 50 cols; 8 waves x 16-row strips.
//     CDNA5 16-bit fragment layout: per lane, K indices are two contiguous
//     8-element runs (base8+0..7 and base8+16..23), so A and B fragments are
//     each TWO ds_load_b128 from bf16-padded LDS panels + a register shuffle.
//     Also accumulates BN sum/sumsq per channel (one atomicAdd per block).
// ---------------------------------------------------------------------------
__global__ void k4_agg(const float* __restrict__ xs, const __bf16* __restrict__ attnT,
                       float* __restrict__ out,
                       float* __restrict__ bnSum, float* __restrict__ bnSumSq) {
    const int n = blockIdx.x / C_;
    const int c = blockIdx.x % C_;
    const int tid = threadIdx.x;          // 256 threads = 8 waves (wave32)
    __shared__ __align__(16) __bf16 Xlb[T_ * VP];   // 16 KB: 128x64 bf16 A panel (zero padded)
    __shared__ __align__(16) __bf16 Blt[VP * VP];   // 8 KB:  64x64 bf16 attn^T (zero padded)
    __shared__ float red[256];

    // stage X (f32 -> bf16, pad cols 50..63 with zero)
    const size_t xbase = (size_t)(n * CT + c * T_) * V_;
    for (int i = tid; i < T_ * VP; i += 256) {
        const int row = i >> 6, col = i & 63;
        const float v = (col < V_) ? xs[xbase + row * V_ + col] : 0.f;
        Xlb[i] = (__bf16)v;
    }
    // stage attn^T (already bf16 + padded) with 16B vector copies
    {
        const uint4* src = (const uint4*)(attnT + (size_t)n * VP * VP);
        uint4* dst = (uint4*)Blt;
        for (int i = tid; i < (VP * VP) / 8; i += 256) dst[i] = src[i];
    }
    __syncthreads();

    const int wv    = tid >> 5;           // wave id 0..7 -> 16-row strip
    const int lane  = tid & 31;
    const int rin   = lane & 15;          // A row / B col / D col within tile
    const int base8 = (lane >> 4) * 8;    // K sub-block select (and D row offset)
    const int arow  = wv * 16 + rin;      // A matrix row within the 128-row panel

    v8f acc[4] = {};
    const __bf16* pa_base = &Xlb[arow * VP + base8];
    #pragma unroll
    for (int kk = 0; kk < 2; ++kk) {      // K = 0..31, 32..63 (padded cols are zero)
        const v8bf a0 = *(const v8bf*)(pa_base + kk * 32);
        const v8bf a1 = *(const v8bf*)(pa_base + kk * 32 + 16);
        const v16bf A = __builtin_shufflevector(a0, a1,
            0, 1, 2, 3, 4, 5, 6, 7, 8, 9, 10, 11, 12, 13, 14, 15);
        #pragma unroll
        for (int ct = 0; ct < 4; ++ct) {  // 4 N-tiles of 16
            const __bf16* pb = &Blt[(ct * 16 + rin) * VP + base8 + kk * 32];
            const v8bf b0 = *(const v8bf*)pb;
            const v8bf b1 = *(const v8bf*)(pb + 16);
            const v16bf B = __builtin_shufflevector(b0, b1,
                0, 1, 2, 3, 4, 5, 6, 7, 8, 9, 10, 11, 12, 13, 14, 15);
            acc[ct] = __builtin_amdgcn_wmma_f32_16x16x32_bf16(
                false, A, false, B, (short)0, acc[ct], false, false);
        }
    }

    // store D (lane = col, VGPR r = row r+base8) + BN partials
    float s1 = 0.f, s2 = 0.f;
    #pragma unroll
    for (int ct = 0; ct < 4; ++ct) {
        const int w = ct * 16 + rin;
        if (w < V_) {
            #pragma unroll
            for (int r = 0; r < 8; ++r) {
                const int t = wv * 16 + base8 + r;
                const float val = acc[ct][r];
                out[(size_t)(n * CT + c * T_ + t) * V_ + w] = val;
                s1 += val; s2 += val * val;
            }
        }
    }
    red[tid] = s1; __syncthreads();
    for (int off = 128; off > 0; off >>= 1) {
        if (tid < off) red[tid] += red[tid + off];
        __syncthreads();
    }
    if (tid == 0) atomicAdd(bnSum + c, red[0]);
    __syncthreads();
    red[tid] = s2; __syncthreads();
    for (int off = 128; off > 0; off >>= 1) {
        if (tid < off) red[tid] += red[tid + off];
        __syncthreads();
    }
    if (tid == 0) atomicAdd(bnSumSq + c, red[0]);
}

// ---------------------------------------------------------------------------
// k5: BatchNorm (batch stats) + affine + ReLU, in place on d_out
// ---------------------------------------------------------------------------
__global__ void k5_bn(float* __restrict__ out, const float* __restrict__ bnSum,
                      const float* __restrict__ bnSumSq,
                      const float* __restrict__ gamma, const float* __restrict__ beta) {
    const int i = blockIdx.x * 256 + threadIdx.x;
    constexpr int total = N_ * C_ * T_ * V_;
    if (i >= total) return;
    const int c = (i / (T_ * V_)) % C_;
    const float invCnt = 1.0f / (float)(N_ * T_ * V_);
    const float m = bnSum[c] * invCnt;
    float var = bnSumSq[c] * invCnt - m * m;
    var = fmaxf(var, 0.f);
    const float y = (out[i] - m) * rsqrtf(var + BN_EPS) * gamma[c] + beta[c];
    out[i] = fmaxf(y, 0.f);
}

// ---------------------------------------------------------------------------
extern "C" void kernel_launch(void* const* d_in, const int* in_sizes, int n_in,
                              void* d_out, int out_size, void* d_ws, size_t ws_size,
                              hipStream_t stream) {
    (void)in_sizes; (void)n_in; (void)out_size; (void)ws_size;
    const float* x     = (const float*)d_in[0];
    const float* Ww    = (const float*)d_in[1];
    const float* Wb    = (const float*)d_in[2];
    const float* att   = (const float*)d_in[3];
    const float* gamma = (const float*)d_in[4];
    const float* beta  = (const float*)d_in[5];
    const int*   mask  = (const int*)d_in[6];
    float* out = (float*)d_out;

    char* ws = (char*)d_ws;
    size_t off = 0;
    float*  xs      = (float*)(ws + off); off += (size_t)N_ * C_ * T_ * V_ * sizeof(float); // 104.9 MB
    float*  gsum    = (float*)(ws + off); off += (size_t)N_ * C_ * V_ * sizeof(float);
    float*  fp      = (float*)(ws + off); off += (size_t)N_ * C_ * V_ * sizeof(float);
    __bf16* attnT   = (__bf16*)(ws + off); off += (size_t)N_ * VP * VP * sizeof(__bf16);    // 16B aligned
    float*  bnSum   = (float*)(ws + off); off += C_ * sizeof(float);
    float*  bnSumSq = (float*)(ws + off); off += C_ * sizeof(float);

    k0_zero  <<<1, 256, 0, stream>>>(bnSum, bnSumSq);
    k1_reduce<<<dim3(N_ * C_), dim3(64, 4), 0, stream>>>(x, xs, gsum);
    k2_fp    <<<N_ * C_, 64, 0, stream>>>(gsum, Ww, Wb, fp);
    k3_attn  <<<N_, 256, 0, stream>>>(fp, att, mask, attnT);
    k4_agg   <<<N_ * C_, 256, 0, stream>>>(xs, attnT, out, bnSum, bnSumSq);
    k5_bn    <<<(N_ * C_ * T_ * V_ + 255) / 256, 256, 0, stream>>>(out, bnSum, bnSumSq, gamma, beta);
}
